// MultiHeadedAttention_29523605192884
// MI455X (gfx1250) — compile-verified, matrix operand order checked by reading, not run on hardware
//
#include <hip/hip_runtime.h>
#include <hip/hip_bf16.h>
#include <math.h>

// ---------------------------------------------------------------- types
typedef __attribute__((ext_vector_type(16))) __bf16 v16bf;
typedef __attribute__((ext_vector_type(8)))  __bf16 v8bf;
typedef __attribute__((ext_vector_type(8)))  float  v8f;
typedef __attribute__((ext_vector_type(4)))  float  v4f;
typedef __attribute__((ext_vector_type(4)))  unsigned int u32x4;
typedef __attribute__((ext_vector_type(8)))  int    i32x8;
typedef __attribute__((ext_vector_type(4)))  int    i32x4;

#define CAT16(lo, hh) __builtin_shufflevector(lo, hh, 0,1,2,3,4,5,6,7,8,9,10,11,12,13,14,15)

__device__ __forceinline__ v8f wmma_bf16(v16bf a, v16bf b, v8f c) {
  return __builtin_amdgcn_wmma_f32_16x16x32_bf16(false, a, false, b, (short)0, c, false, false);
}

// Load a 16-bf16 A/B-operand fragment from K-contiguous storage.
// Per ISA 7.12.2 (16-bit 16x32): lane<16 -> K chunks [0..7],[16..23];
// lane>=16 -> [8..15],[24..31]; row = lane&15.
__device__ __forceinline__ v16bf frag_ld(const __bf16* rowp, int hi) {
  v8bf lo = *(const v8bf*)(rowp + hi * 8);
  v8bf hh = *(const v8bf*)(rowp + 16 + hi * 8);
  return CAT16(lo, hh);
}

// ---- CDNA5 async global->LDS copy (ASYNCcnt-tracked, bypasses VGPRs) ----
__device__ __forceinline__ void async_b128(void* lds_ptr, const void* base, int byte_off) {
  unsigned lds_off = (unsigned)(uintptr_t)lds_ptr;
  asm volatile("global_load_async_to_lds_b128 %0, %1, %2"
               :: "v"(lds_off), "v"(byte_off), "s"(base)
               : "memory");
}
__device__ __forceinline__ void wait_async0() {
  asm volatile("s_wait_asynccnt 0x0" ::: "memory");
}

// ---- CDNA5 Tensor Data Mover: 2D bf16 tile, global -> LDS ----
// D# packed per CDNA5 ISA (group0 128b, group1 256b).
// tile_w elems per row, tile_h rows, row_stride in elements, 2B/elem.
// This toolchain's builtin takes 6 args: (g0, g1, g2, g3, extra, cpol).
__device__ __forceinline__ void tdm_load_2d(const void* gaddr, unsigned lds_off,
                                            unsigned tile_w, unsigned tile_h,
                                            unsigned row_stride) {
  unsigned long long ga = (unsigned long long)(uintptr_t)gaddr;
  u32x4 g0;
  g0[0] = 1u;                                           // count=1, user descriptor
  g0[1] = lds_off;                                      // lds_addr [63:32]
  g0[2] = (unsigned)(ga & 0xFFFFFFFFu);                 // global_addr lo
  g0[3] = (unsigned)((ga >> 32) & 0x1FFFFFFu)           // global_addr [56:32]
        | (2u << 30);                                   // type=2 ("image")
  i32x8 g1;
  g1[0] = (int)(1u << 16);                              // data_size=1 (2B); mask=0; no pad
  g1[1] = (int)((tile_w & 0xFFFFu) << 16);              // tensor_dim0 lo16  [63:48]
  g1[2] = (int)((tile_h & 0xFFFFu) << 16);              // tensor_dim1 lo16  [95:80]
  g1[3] = (int)((tile_w & 0xFFFFu) << 16);              // tile_dim0         [127:112]
  g1[4] = (int)(tile_h & 0xFFFFu);                      // tile_dim1         [143:128]
  g1[5] = (int)row_stride;                              // tensor_dim0_stride lo32
  g1[6] = 0;
  g1[7] = 0;
  i32x4 z4 = {0, 0, 0, 0};                              // groups 2/3 unused (2D)
  i32x8 z8 = {0, 0, 0, 0, 0, 0, 0, 0};
  __builtin_amdgcn_tensor_load_to_lds(g0, g1, z4, z4, z8, 0);
}

// ---------------------------------------------------------------- problem dims
#define S_LEN  2048
#define BATCH  2
#define DM     1024
#define NH     16
#define DK     64
#define MROWS  (BATCH * S_LEN)   // 4096

// ---------------------------------------------------------------- fp32 -> bf16
__global__ void cvt_f32_bf16(const float* __restrict__ src, __bf16* __restrict__ dst, int n) {
  int i = (blockIdx.x * blockDim.x + threadIdx.x) * 8;
  if (i + 8 <= n) {
    v4f a = *(const v4f*)(src + i);
    v4f b = *(const v4f*)(src + i + 4);
    v8bf o;
#pragma unroll
    for (int e = 0; e < 4; ++e) { o[e] = (__bf16)a[e]; o[4 + e] = (__bf16)b[e]; }
    *(v8bf*)(dst + i) = o;
  }
}

// ---------------------------------------------------------------- GEMM (NT): C[M,N] = A[M,K] * W[N,K]^T + bias
// bf16 operands; block tile 128x128, BK=32; 8 waves x (64x32).
// Double-buffered LDS, async-staged: loads for tile t+1 overlap WMMAs of tile t.
template <typename OT>
__global__ void __launch_bounds__(256)
gemm_wmma(const __bf16* __restrict__ A, const __bf16* __restrict__ W,
          const float* __restrict__ bias, OT* __restrict__ C,
          int M, int N, int K) {
  constexpr int BK  = 32;
  constexpr int LDA = 48;               // padded LDS row stride (96B, 16B multiple)
  __shared__ __bf16 As[2][128 * LDA];
  __shared__ __bf16 Bs[2][128 * LDA];

  const int tid  = threadIdx.x;
  const int lane = tid & 31;
  const int wave = tid >> 5;
  const int wm   = wave >> 2;           // 0..1 : 64-row slab
  const int wn   = wave & 3;            // 0..3 : 32-col slab
  const int lr   = lane & 15;
  const int hi   = lane >> 4;
  const int m0   = blockIdx.y * 128;
  const int n0   = blockIdx.x * 128;

  // global->LDS mapping: thread stages 16 consecutive bf16 of one tile row
  const int ldr = tid >> 1;
  const int ldc = (tid & 1) * 16;

  auto stage = [&](int kk, int buf) {
    const int a_off = ((m0 + ldr) * K + kk + ldc) * 2;  // bytes
    const int b_off = ((n0 + ldr) * K + kk + ldc) * 2;  // bytes
    __bf16* ad = &As[buf][ldr * LDA + ldc];
    __bf16* bd = &Bs[buf][ldr * LDA + ldc];
    async_b128(ad,     A, a_off);
    async_b128(ad + 8, A, a_off + 16);
    async_b128(bd,     W, b_off);
    async_b128(bd + 8, W, b_off + 16);
  };

  v8f acc[4][2];
#pragma unroll
  for (int i = 0; i < 4; ++i)
#pragma unroll
    for (int j = 0; j < 2; ++j) acc[i][j] = (v8f){};

  // prologue: stage tile 0
  stage(0, 0);
  wait_async0();
  __syncthreads();

  int cur = 0;
  for (int k0 = 0; k0 < K; k0 += BK) {
    const int nxt = cur ^ 1;
    if (k0 + BK < K) stage(k0 + BK, nxt);   // async loads fly during compute

    v16bf af[4], bfr[2];
#pragma unroll
    for (int i = 0; i < 4; ++i)
      af[i] = frag_ld(&As[cur][(wm * 64 + 16 * i + lr) * LDA], hi);
#pragma unroll
    for (int j = 0; j < 2; ++j)
      bfr[j] = frag_ld(&Bs[cur][(wn * 32 + 16 * j + lr) * LDA], hi);
#pragma unroll
    for (int i = 0; i < 4; ++i)
#pragma unroll
      for (int j = 0; j < 2; ++j)
        acc[i][j] = wmma_bf16(af[i], bfr[j], acc[i][j]);

    wait_async0();          // next tile landed (latency hidden behind WMMAs)
    __syncthreads();
    cur = nxt;
  }

  // ---- epilogue: +bias, store
#pragma unroll
  for (int i = 0; i < 4; ++i) {
#pragma unroll
    for (int j = 0; j < 2; ++j) {
      const int colg = n0 + wn * 32 + 16 * j + lr;
      const float bv = bias[colg];
#pragma unroll
      for (int r = 0; r < 8; ++r) {
        const int rowg = m0 + wm * 64 + 16 * i + r + 8 * hi;
        const float v = acc[i][j][r] + bv;
        if constexpr (sizeof(OT) == 4) C[(size_t)rowg * N + colg] = v;
        else                           C[(size_t)rowg * N + colg] = (__bf16)v;
      }
    }
  }
}

// ---------------------------------------------------------------- flash attention
// grid: (S/128, H, B); 8 waves/block, each wave owns a 16-row Q tile.
// K tiles staged by the Tensor Data Mover (TENSORcnt), V transposed manually,
// both double-buffered and pipelined against the WMMA/softmax work.
__global__ void __launch_bounds__(256)
flash_attn(const __bf16* __restrict__ Q, const __bf16* __restrict__ Kb,
           const __bf16* __restrict__ Vb, __bf16* __restrict__ Ctx) {
  __shared__ __bf16 Ks[2][32 * 64];     // [key][d]   (TDM-staged)
  __shared__ __bf16 Vt[2][64 * 32];     // [d][key]   (manual transpose)
  __shared__ __bf16 Ps[8 * 16 * 32];    // per-wave P scratch [16][32]

  const int tid  = threadIdx.x;
  const int lane = tid & 31;
  const int wave = tid >> 5;
  const int lr   = lane & 15;
  const int hi   = lane >> 4;
  const int h    = blockIdx.y;
  const int b    = blockIdx.z;
  const int q0   = blockIdx.x * 128 + wave * 16;
  const int rowbase = b * S_LEN;
  const int hcol = h * DK;

  // Q fragments for this wave (k-steps over d = 0..63), kept in registers.
  v16bf aq[2];
#pragma unroll
  for (int ks = 0; ks < 2; ++ks)
    aq[ks] = frag_ld(Q + (size_t)(rowbase + q0 + lr) * DM + hcol + ks * 32, hi);

  v8f o[4];
#pragma unroll
  for (int j = 0; j < 4; ++j) o[j] = (v8f){};
  float rmax[8], rsum[8];
#pragma unroll
  for (int r = 0; r < 8; ++r) { rmax[r] = -INFINITY; rsum[r] = 0.f; }

  const float sm_scale = 0.125f;        // 1/sqrt(64)
  const int kr = tid >> 3;              // 0..31 key row
  const int kc = (tid & 7) * 8;         // 8-elem chunk of d

  auto stage = [&](int kt, int buf) {
    // V tile: load + transpose through VGPRs
    v8bf v = *(const v8bf*)(Vb + (size_t)(rowbase + kt + kr) * DM + hcol + kc);
#pragma unroll
    for (int e = 0; e < 8; ++e) Vt[buf][(kc + e) * 32 + kr] = v[e];
    // K tile: one TDM descriptor moves the whole 32x64 tile
    if (wave == 0)
      tdm_load_2d(Kb + (size_t)(rowbase + kt) * DM + hcol,
                  (unsigned)(uintptr_t)&Ks[buf][0], DK, 32, DM);
  };

  // prologue: stage tile 0
  stage(0, 0);
  if (wave == 0) __builtin_amdgcn_s_wait_tensorcnt(0);
  __syncthreads();

  int cur = 0;
  for (int kt = 0; kt < S_LEN; kt += 32) {
    const int nxt = cur ^ 1;
    if (kt + 32 < S_LEN) stage(kt + 32, nxt);   // TDM + V copy fly during compute

    // ---- scores: S(16x32) = Q(16x64) @ K^T, two 16x16 N-tiles, K-dim 64 = 2 steps
    v8f s[2] = {};
#pragma unroll
    for (int j = 0; j < 2; ++j)
#pragma unroll
      for (int ks = 0; ks < 2; ++ks)
        s[j] = wmma_bf16(aq[ks], frag_ld(&Ks[cur][(16 * j + lr) * 64 + ks * 32], hi), s[j]);

    // ---- online softmax (rows live across 16 lanes; xor-shuffle reduce)
    __bf16* pw = Ps + wave * 512;
#pragma unroll
    for (int r = 0; r < 8; ++r) {
      float s0 = s[0][r] * sm_scale, s1 = s[1][r] * sm_scale;
      float rm = fmaxf(s0, s1);
#pragma unroll
      for (int off = 1; off < 16; off <<= 1) rm = fmaxf(rm, __shfl_xor(rm, off, 32));
      const float nm  = fmaxf(rmax[r], rm);
      const float scl = __expf(rmax[r] - nm);
      const float p0  = __expf(s0 - nm);
      const float p1  = __expf(s1 - nm);
      float rs = p0 + p1;
#pragma unroll
      for (int off = 1; off < 16; off <<= 1) rs += __shfl_xor(rs, off, 32);
      rsum[r] = rsum[r] * scl + rs;
      rmax[r] = nm;
#pragma unroll
      for (int j = 0; j < 4; ++j) o[j][r] *= scl;
      // spill P in C-layout so it can be re-read in A-operand layout
      pw[(r + 8 * hi) * 32 + lr]      = (__bf16)p0;
      pw[(r + 8 * hi) * 32 + 16 + lr] = (__bf16)p1;
    }
    // same-wave LDS RAW: make P stores visible before fragment re-load
    asm volatile("s_wait_dscnt 0x0" ::: "memory");

    // ---- O += P(16x32) @ V(32x64): 4 d-tiles, single K-step of 32
    v16bf ap = frag_ld(pw + lr * 32, hi);
#pragma unroll
    for (int j = 0; j < 4; ++j)
      o[j] = wmma_bf16(ap, frag_ld(&Vt[cur][(16 * j + lr) * 32], hi), o[j]);

    if (wave == 0) __builtin_amdgcn_s_wait_tensorcnt(0);
    __syncthreads();
    cur = nxt;
  }

  // ---- normalize + store ctx (bf16, [s][h*64+d] layout)
#pragma unroll
  for (int r = 0; r < 8; ++r) {
    const float inv = 1.0f / rsum[r];
    const size_t base = (size_t)(rowbase + q0 + r + 8 * hi) * DM + hcol;
#pragma unroll
    for (int j = 0; j < 4; ++j)
      Ctx[base + 16 * j + lr] = (__bf16)(o[j][r] * inv);
  }
}

// ---------------------------------------------------------------- launcher
extern "C" void kernel_launch(void* const* d_in, const int* in_sizes, int n_in,
                              void* d_out, int out_size, void* d_ws, size_t ws_size,
                              hipStream_t stream) {
  const float* query = (const float*)d_in[0];
  const float* key_  = (const float*)d_in[1];
  const float* value = (const float*)d_in[2];
  const float* Wq = (const float*)d_in[3];
  const float* bq = (const float*)d_in[4];
  const float* Wk = (const float*)d_in[5];
  const float* bk = (const float*)d_in[6];
  const float* Wv = (const float*)d_in[7];
  const float* bv = (const float*)d_in[8];
  const float* Wo = (const float*)d_in[9];
  const float* bo = (const float*)d_in[10];
  float* out = (float*)d_out;

  const size_t WSZ = (size_t)DM * DM;       // 1M elems
  const size_t MSZ = (size_t)MROWS * DM;    // 4M elems
  __bf16* Wq_b = (__bf16*)d_ws;
  __bf16* Wk_b = Wq_b + WSZ;
  __bf16* Wv_b = Wk_b + WSZ;
  __bf16* Wo_b = Wv_b + WSZ;
  __bf16* Xq_b = Wo_b + WSZ;                // bf16 activations
  __bf16* Xk_b = Xq_b + MSZ;
  __bf16* Xv_b = Xk_b + MSZ;
  __bf16* Qb   = Xv_b + MSZ;
  __bf16* Kb   = Qb + MSZ;
  __bf16* Vb   = Kb + MSZ;
  __bf16* Cb   = Vb + MSZ;                  // total 32M bf16 = 64 MB

  const int wBlk = (int)((WSZ / 8 + 255) / 256);
  const int mBlk = (int)((MSZ / 8 + 255) / 256);
  cvt_f32_bf16<<<wBlk, 256, 0, stream>>>(Wq, Wq_b, (int)WSZ);
  cvt_f32_bf16<<<wBlk, 256, 0, stream>>>(Wk, Wk_b, (int)WSZ);
  cvt_f32_bf16<<<wBlk, 256, 0, stream>>>(Wv, Wv_b, (int)WSZ);
  cvt_f32_bf16<<<wBlk, 256, 0, stream>>>(Wo, Wo_b, (int)WSZ);
  cvt_f32_bf16<<<mBlk, 256, 0, stream>>>(query, Xq_b, (int)MSZ);
  cvt_f32_bf16<<<mBlk, 256, 0, stream>>>(key_,  Xk_b, (int)MSZ);
  cvt_f32_bf16<<<mBlk, 256, 0, stream>>>(value, Xv_b, (int)MSZ);

  dim3 ggrd(DM / 128, MROWS / 128);
  gemm_wmma<__bf16><<<ggrd, 256, 0, stream>>>(Xq_b, Wq_b, bq, Qb, MROWS, DM, DM);
  gemm_wmma<__bf16><<<ggrd, 256, 0, stream>>>(Xk_b, Wk_b, bk, Kb, MROWS, DM, DM);
  gemm_wmma<__bf16><<<ggrd, 256, 0, stream>>>(Xv_b, Wv_b, bv, Vb, MROWS, DM, DM);

  dim3 fgrd(S_LEN / 128, NH, BATCH);
  flash_attn<<<fgrd, 256, 0, stream>>>(Qb, Kb, Vb, Cb);

  gemm_wmma<float><<<ggrd, 256, 0, stream>>>(Cb, Wo_b, bo, out, MROWS, DM, DM);
}
